// PointEncoder_21354577396094
// MI455X (gfx1250) — matrix operand order repaired and checked by reading, not compiled, not run on hardware
//
#include <hip/hip_runtime.h>
#include <math.h>

// ---------------- problem constants (from reference) ----------------
#define BATCH 2
#define NPTS  4096
#define KNB   40      // neighbors
#define RANK  32
#define HID   64

typedef __attribute__((ext_vector_type(16))) _Float16 v16h;
typedef __attribute__((ext_vector_type(8)))  _Float16 v8h;
typedef __attribute__((ext_vector_type(8)))  float    v8f;
typedef __attribute__((ext_vector_type(4)))  float    v4f;

// ---------------- WMMA fragment builders (contiguous-k order, same in A and B) ----------------
// A-matrix 16x32 f16, activations stored row-major [row][k] in LDS.
// Lane L: row = L&15, g = L>>4. Elements 0..15 hold K = 16g+0..15
// (contiguous) -> two ds_load_b128.
__device__ __forceinline__ v16h ldsA16x32(const _Float16* base, int ld, int kbase) {
  const int lane = threadIdx.x & 31;
  const int row  = lane & 15;
  const int g    = lane >> 4;
  const _Float16* p = base + row * ld + kbase + 16 * g;
  const v8h lo = *(const v8h*)(p);
  const v8h hi = *(const v8h*)(p + 8);
  v16h a;
#pragma unroll
  for (int h = 0; h < 8; ++h) { a[h] = lo[h]; a[h + 8] = hi[h]; }
  return a;
}

// B-matrix 32x16 f16 (KxN), weights stored TRANSPOSED [n][k] in LDS.
// Lane L: col = (L&15)+nbase, g = L>>4. Elements 0..15 hold K = kbase+16g+0..15
// (contiguous along k) -> two ds_load_b128.
__device__ __forceinline__ v16h ldsBT32x16(const _Float16* baseT, int ldk, int kbase, int nbase) {
  const int lane = threadIdx.x & 31;
  const int col  = (lane & 15) + nbase;
  const int g    = lane >> 4;
  const _Float16* p = baseT + col * ldk + kbase + 16 * g;
  const v8h lo = *(const v8h*)(p);
  const v8h hi = *(const v8h*)(p + 8);
  v16h b;
#pragma unroll
  for (int h = 0; h < 8; ++h) { b[h] = lo[h]; b[h + 8] = hi[h]; }
  return b;
}

__device__ __forceinline__ v8f wmma_f16(v16h a, v16h b, v8f c) {
  return __builtin_amdgcn_wmma_f32_16x16x32_f16(false, a, false, b, (short)0, c, false, false);
}

// C/D 16x16 f32: lane = col (lane&15), VGPR r -> row r + (lane>=16 ? 8 : 0)

// ---------------- monotone float<->uint for atomic max ----------------
__device__ __forceinline__ unsigned int fkey(float f) {
  unsigned int u = __float_as_uint(f);
  return (u & 0x80000000u) ? ~u : (u | 0x80000000u);
}
__device__ __forceinline__ float fdec(unsigned int u) {
  unsigned int b = (u & 0x80000000u) ? (u & 0x7fffffffu) : ~u;
  return __uint_as_float(b);
}

// ---------------- kernel 0: zero accumulators ----------------
__global__ void pe_init_kernel(float* contracted, unsigned int* glob, int nc) {
  int i = blockIdx.x * 256 + threadIdx.x;
  if (i < nc) contracted[i] = 0.f;
  if (i < BATCH * 8) glob[i] = 0u;   // key 0 < key of any real float
}

// ---------------- kernel 1: streaming top-K (smallest distances) ----------------
// one 256-thread block per (b,n) row; 16 values/thread (4 NT float4 loads);
// 40 argmin passes; wave32 shfl_xor butterfly + 8-entry cross-wave LDS min.
__global__ __launch_bounds__(256) void pe_topk_kernel(const float* __restrict__ dist,
                                                      int* __restrict__ nbrs) {
  __shared__ unsigned long long swk[8];
  const int tid  = threadIdx.x;
  const int lane = tid & 31;
  const int wave = tid >> 5;
  const size_t row = blockIdx.x;                 // b*NPTS + n
  const float* dr = dist + row * NPTS;

  // element (c,j): global index = c*1024 + tid*4 + j
  float vals[16];
#pragma unroll
  for (int c = 0; c < 4; ++c) {
    const v4f t = __builtin_nontemporal_load((const v4f*)(dr + c * 1024 + tid * 4));
#pragma unroll
    for (int j = 0; j < 4; ++j) vals[c * 4 + j] = t[j];
  }

  for (int it = 0; it < KNB; ++it) {
    // thread-local argmin (strict < keeps lowest index on ties)
    float bv = vals[0]; int bi = 0;
#pragma unroll
    for (int i = 1; i < 16; ++i) if (vals[i] < bv) { bv = vals[i]; bi = i; }
    const unsigned myidx = (unsigned)((bi >> 2) * 1024 + tid * 4 + (bi & 3));
    // pack: nonneg-float bits are order-isomorphic to uint; low 32 = index (tie -> low idx)
    unsigned long long key = ((unsigned long long)__float_as_uint(bv) << 32) | myidx;
    // wave32 butterfly min
#pragma unroll
    for (int off = 16; off > 0; off >>= 1) {
      const unsigned long long o = __shfl_xor(key, off, 32);
      if (o < key) key = o;
    }
    if (lane == 0) swk[wave] = key;
    __syncthreads();
    unsigned long long w = swk[0];
#pragma unroll
    for (int q = 1; q < 8; ++q) { const unsigned long long o = swk[q]; if (o < w) w = o; }
    const unsigned widx = (unsigned)(w & 0xffffffffu);
    if (tid == 0) nbrs[row * KNB + it] = (int)widx;
    // mask the winner with constant-index compares (vals stays in VGPRs)
#pragma unroll
    for (int i = 0; i < 16; ++i) {
      const unsigned ii = (unsigned)((i >> 2) * 1024 + tid * 4 + (i & 3));
      if (ii == widx) vals[i] = __builtin_inff();
    }
    __syncthreads();   // protect swk before next iteration
  }
}

// ---------------- kernel 2: gather + rel + kernel-MLP via WMMA ----------------
// 8 waves/block, each wave owns a 16-row tile of the [B*N*K, *] activation matrix.
__global__ __launch_bounds__(256) void pe_mlp_kernel(
    const float* __restrict__ pc, const int* __restrict__ nbrs,
    const float* __restrict__ W1, const float* __restrict__ b1,
    const float* __restrict__ W2, const float* __restrict__ b2,
    const float* __restrict__ W3, const float* __restrict__ b3,
    float* __restrict__ contracted) {
  // weights stored transposed [n][k] for vector B-fragment loads
  __shared__ alignas(16) _Float16 sW1t[HID * 32];    // [64][32], k padded 3->32
  __shared__ alignas(16) _Float16 sW2t[HID * HID];   // [64][64]
  __shared__ alignas(16) _Float16 sW3t[RANK * HID];  // [32][64]
  __shared__ float    sb1[HID], sb2[HID], sb3[RANK];
  __shared__ alignas(16) _Float16 sX [8][16 * 32];   // rel, K-padded
  __shared__ alignas(16) _Float16 sH1[8][16 * HID];
  __shared__ alignas(16) _Float16 sH2[8][16 * HID];
  __shared__ float    sNrm[8][16];

  const int tid = threadIdx.x;
  for (int i = tid; i < HID * 32; i += 256) {        // sW1t[n*32+k] = W1[k][n]
    const int n = i >> 5, k = i & 31;
    sW1t[i] = (_Float16)((k < 3) ? W1[k * HID + n] : 0.f);
  }
  for (int i = tid; i < HID * HID; i += 256) {       // sW2t[n*64+k] = W2[k][n]
    const int n = i >> 6, k = i & 63;
    sW2t[i] = (_Float16)W2[k * HID + n];
  }
  for (int i = tid; i < RANK * HID; i += 256) {      // sW3t[n*64+k] = W3[k][n]
    const int n = i >> 6, k = i & 63;
    sW3t[i] = (_Float16)W3[k * RANK + n];
  }
  if (tid < HID)  { sb1[tid] = b1[tid]; sb2[tid] = b2[tid]; }
  if (tid < RANK) sb3[tid] = b3[tid];

  const int wave = tid >> 5;
  const int lane = tid & 31;
  const long rowbase = ((long)blockIdx.x * 8 + wave) * 16;

  // build rel rows (lanes 0..15 own the 16 rows of this wave's tile)
  if (lane < 16) {
    const long r  = rowbase + lane;
    const int bb  = (int)(r / ((long)NPTS * KNB));
    const int rem = (int)(r % ((long)NPTS * KNB));
    const int n   = rem / KNB;
    const int k   = rem % KNB;
    const int idx = nbrs[((long)(bb * NPTS + n)) * KNB + k];
    const float cx = pc[((long)(bb * NPTS + n)) * 3 + 0];
    const float cy = pc[((long)(bb * NPTS + n)) * 3 + 1];
    const float cz = pc[((long)(bb * NPTS + n)) * 3 + 2];
    const float rx = pc[((long)(bb * NPTS) + idx) * 3 + 0] - cx;
    const float ry = pc[((long)(bb * NPTS) + idx) * 3 + 1] - cy;
    const float rz = pc[((long)(bb * NPTS) + idx) * 3 + 2] - cz;
    sNrm[wave][lane] = sqrtf(rx * rx + ry * ry + rz * rz);
    _Float16* xr = &sX[wave][lane * 32];
    v8h x0 = {};
    x0[0] = (_Float16)rx; x0[1] = (_Float16)ry; x0[2] = (_Float16)rz;
    const v8h z = {};
    *(v8h*)(xr)      = x0;
    *(v8h*)(xr + 8)  = z;
    *(v8h*)(xr + 16) = z;
    *(v8h*)(xr + 24) = z;
  }
  __syncthreads();

  const int colq = lane & 15;
  const int mb   = (lane >> 4) * 8;

  // ---- layer 1: [16x32] x [32x64] ----
  {
    v16h a = ldsA16x32(sX[wave], 32, 0);
#pragma unroll
    for (int t = 0; t < 4; ++t) {
      v8f c = {};
      c = wmma_f16(a, ldsBT32x16(sW1t, 32, 0, t * 16), c);
      const int col = colq + t * 16;
#pragma unroll
      for (int r2 = 0; r2 < 8; ++r2) {
        float v = c[r2] + sb1[col];
        v = v > 0.f ? v : 0.f;
        sH1[wave][(mb + r2) * HID + col] = (_Float16)v;
      }
    }
  }
  __syncthreads();

  // ---- layer 2: [16x64] x [64x64] ----
  {
    v16h a0 = ldsA16x32(sH1[wave], HID, 0);
    v16h a1 = ldsA16x32(sH1[wave], HID, 32);
#pragma unroll
    for (int t = 0; t < 4; ++t) {
      v8f c = {};
      c = wmma_f16(a0, ldsBT32x16(sW2t, HID, 0,  t * 16), c);
      c = wmma_f16(a1, ldsBT32x16(sW2t, HID, 32, t * 16), c);
      const int col = colq + t * 16;
#pragma unroll
      for (int r2 = 0; r2 < 8; ++r2) {
        float v = c[r2] + sb2[col];
        v = v > 0.f ? v : 0.f;
        sH2[wave][(mb + r2) * HID + col] = (_Float16)v;
      }
    }
  }
  __syncthreads();

  // ---- layer 3: [16x64] x [64x32], scale by ||rel||/K, atomic-accumulate over K ----
  {
    v16h a0 = ldsA16x32(sH2[wave], HID, 0);
    v16h a1 = ldsA16x32(sH2[wave], HID, 32);
#pragma unroll
    for (int t = 0; t < 2; ++t) {
      v8f c = {};
      c = wmma_f16(a0, ldsBT32x16(sW3t, HID, 0,  t * 16), c);
      c = wmma_f16(a1, ldsBT32x16(sW3t, HID, 32, t * 16), c);
      const int col = colq + t * 16;
#pragma unroll
      for (int r2 = 0; r2 < 8; ++r2) {
        const long r  = rowbase + mb + r2;
        const int bb  = (int)(r / ((long)NPTS * KNB));
        const int rem = (int)(r % ((long)NPTS * KNB));
        const int n   = rem / KNB;
        const float v = (c[r2] + sb3[col]) * sNrm[wave][mb + r2] * (1.0f / KNB);
        atomicAdd(&contracted[((long)(bb * NPTS + n)) * RANK + col], v);
      }
    }
  }
}

// ---------------- kernel 3: y = relu(contracted@Wo+bo); tran=y@Wg+bg; global max ----
__global__ __launch_bounds__(256) void pe_head_kernel(
    const float* __restrict__ contracted,
    const float* __restrict__ Wo, const float* __restrict__ bo,
    const float* __restrict__ Wg, const float* __restrict__ bg,
    float* __restrict__ out, unsigned int* __restrict__ glob) {
  __shared__ alignas(16) _Float16 sWot[32 * 32];   // [n][k]
  __shared__ alignas(16) _Float16 sWgt[16 * 32];   // [n][k], n padded 8->16
  __shared__ float    sbo[32], sbg[8];
  __shared__ alignas(16) _Float16 sX[8][16 * 32];
  __shared__ alignas(16) _Float16 sY[8][16 * 32];

  const int tid = threadIdx.x;
  for (int i = tid; i < 32 * 32; i += 256) {       // sWot[n*32+k] = Wo[k][n]
    const int n = i >> 5, k = i & 31;
    sWot[i] = (_Float16)Wo[k * 32 + n];
  }
  for (int i = tid; i < 16 * 32; i += 256) {       // sWgt[n*32+k] = Wg[k][n] (n<8)
    const int n = i >> 5, k = i & 31;
    sWgt[i] = (_Float16)((n < 8) ? Wg[k * 8 + n] : 0.f);
  }
  if (tid < 32) sbo[tid] = bo[tid];
  if (tid < 8)  sbg[tid] = bg[tid];

  const int wave = tid >> 5;
  const int lane = tid & 31;
  const long rowbase = ((long)blockIdx.x * 8 + wave) * 16;   // rows over B*N

  for (int i = lane; i < 16 * 32; i += 32) {
    const int rr = i >> 5, cc = i & 31;
    sX[wave][i] = (_Float16)contracted[(rowbase + rr) * 32 + cc];
  }
  __syncthreads();

  const int colq = lane & 15;
  const int mb   = (lane >> 4) * 8;

  // y = relu(X @ Wo + bo); write f32 to out[:,0:32] straight from D layout
  {
    v16h a = ldsA16x32(sX[wave], 32, 0);
#pragma unroll
    for (int t = 0; t < 2; ++t) {
      v8f c = {};
      c = wmma_f16(a, ldsBT32x16(sWot, 32, 0, t * 16), c);
      const int col = colq + t * 16;
#pragma unroll
      for (int r2 = 0; r2 < 8; ++r2) {
        float v = c[r2] + sbo[col];
        v = v > 0.f ? v : 0.f;
        out[(rowbase + mb + r2) * 40 + col] = v;
        sY[wave][(mb + r2) * 32 + col] = (_Float16)v;
      }
    }
  }
  __syncthreads();

  // tran = y @ Wg + bg; fold into per-batch global max
  {
    v16h a = ldsA16x32(sY[wave], 32, 0);
    v8f c = {};
    c = wmma_f16(a, ldsBT32x16(sWgt, 32, 0, 0), c);
    if (colq < 8) {
#pragma unroll
      for (int r2 = 0; r2 < 8; ++r2) {
        const long r = rowbase + mb + r2;
        const int bb = (int)(r / NPTS);
        const float v = c[r2] + sbg[colq];
        atomicMax(&glob[bb * 8 + colq], fkey(v));
      }
    }
  }
}

// ---------------- kernel 4: broadcast global max into out[:,32:40] ----------------
__global__ void pe_bcast_kernel(const unsigned int* __restrict__ glob,
                                float* __restrict__ out) {
  const int i = blockIdx.x * 256 + threadIdx.x;
  if (i >= BATCH * NPTS * 8) return;
  const int j = i & 7;
  const long r = i >> 3;
  const int bb = (int)(r / NPTS);
  out[r * 40 + 32 + j] = fdec(glob[bb * 8 + j]);
}

// ---------------- launcher ----------------
extern "C" void kernel_launch(void* const* d_in, const int* in_sizes, int n_in,
                              void* d_out, int out_size, void* d_ws, size_t ws_size,
                              hipStream_t stream) {
  const float* pc   = (const float*)d_in[0];
  const float* dist = (const float*)d_in[1];
  const float* W1 = (const float*)d_in[2];  const float* b1 = (const float*)d_in[3];
  const float* W2 = (const float*)d_in[4];  const float* b2 = (const float*)d_in[5];
  const float* W3 = (const float*)d_in[6];  const float* b3 = (const float*)d_in[7];
  const float* Wo = (const float*)d_in[8];  const float* bo = (const float*)d_in[9];
  const float* Wg = (const float*)d_in[10]; const float* bg = (const float*)d_in[11];
  float* out = (float*)d_out;

  char* ws = (char*)d_ws;
  float*        contracted = (float*)ws;                                  // B*N*32 f32
  int*          nbrs = (int*)(ws + (size_t)BATCH * NPTS * RANK * 4);      // B*N*K  i32
  unsigned int* glob = (unsigned int*)(ws + (size_t)BATCH * NPTS * RANK * 4
                                          + (size_t)BATCH * NPTS * KNB * 4); // B*8 u32

  const int nc = BATCH * NPTS * RANK;
  pe_init_kernel<<<(nc + 255) / 256, 256, 0, stream>>>(contracted, glob, nc);

  pe_topk_kernel<<<BATCH * NPTS, 256, 0, stream>>>(dist, nbrs);

  pe_mlp_kernel<<<(BATCH * NPTS * KNB) / (16 * 8), 256, 0, stream>>>(
      pc, nbrs, W1, b1, W2, b2, W3, b3, contracted);

  pe_head_kernel<<<(BATCH * NPTS) / (16 * 8), 256, 0, stream>>>(
      contracted, Wo, bo, Wg, bg, out, glob);

  pe_bcast_kernel<<<(BATCH * NPTS * 8 + 255) / 256, 256, 0, stream>>>(glob, out);
}
